// SelfAttention_46273977647822
// MI455X (gfx1250) — matrix operand order repaired and checked by reading, not compiled, not run on hardware
//
#include <hip/hip_runtime.h>
#include <hip/hip_fp16.h>

// ---------------------------------------------------------------------------
// Self-attention (B=4, N=4096, C=512, Dqk=64) for gfx1250.
// GEMMs on v_wmma_f32_16x16x32_f16; key tiles double-buffered into LDS via
// global_load_async_to_lds_b128 (ASYNCcnt-pipelined); flash-attention style.
// ---------------------------------------------------------------------------

typedef __attribute__((ext_vector_type(16))) _Float16 v16h;
typedef __attribute__((ext_vector_type(8)))  _Float16 v8h;
typedef __attribute__((ext_vector_type(8)))  float    v8f;

#define LOG2E 1.4426950408889634f

__device__ __forceinline__ v8f zero_v8f() {
  v8f z;
#pragma unroll
  for (int i = 0; i < 8; ++i) z[i] = 0.f;
  return z;
}

__device__ __forceinline__ v8f wmma_f16(v16h a, v16h b, v8f c) {
  // (neg_a, A, neg_b, B, c_mod, C, reuse_a, reuse_b)
  return __builtin_amdgcn_wmma_f32_16x16x32_f16(false, a, false, b, (short)0, c,
                                                false, false);
}

// Generic->LDS: for the LDS aperture the low 32 bits of a generic pointer ARE
// the LDS byte address (ISA 10.2).
__device__ __forceinline__ uint32_t lds_addr32(const void* p) {
  return (uint32_t)(uintptr_t)p;
}

// Async-copy a contiguous 4KB block (g key-tile: 32 rows x 64 f16) into LDS.
// 128 threads x 2 x b128. Tracked by ASYNCcnt; waited with s_wait_asynccnt.
__device__ __forceinline__ void async_copy_4k(const _Float16* __restrict__ src,
                                              const _Float16* dst_lds, int tid) {
  const uint64_t s64 = (uint64_t)(uintptr_t)src;
  const uint32_t base = lds_addr32(dst_lds);
#pragma unroll
  for (int i = 0; i < 2; ++i) {
    const uint32_t off = (uint32_t)(tid + i * 128) * 16u;
    const uint32_t l = base + off;
    asm volatile("global_load_async_to_lds_b128 %0, %1, %2"
                 :: "v"(l), "v"(off), "s"(s64) : "memory");
  }
}

// ---- A-operand (16x32, M x K) loaders -------------------------------------
// Layout: lanes 0-15 hold M=lane; halves 0-15 -> K 0-15.
//         lanes 16-31 hold M=lane-16; halves 0-15 -> K 16-31.
__device__ __forceinline__ void load_a_f32(v16h& a, const float* __restrict__ X,
                                           int ld, int rowBase, int kBase, int lane) {
  const int m  = lane & 15;
  const int hi = lane >> 4;
  const float* row = X + (size_t)(rowBase + m) * ld + kBase;
  const int k0 = hi ? 16 : 0;
  const int k1 = hi ? 24 : 8;
#pragma unroll
  for (int i = 0; i < 8; ++i) a[i] = (_Float16)row[k0 + i];
#pragma unroll
  for (int i = 0; i < 8; ++i) a[8 + i] = (_Float16)row[k1 + i];
}

__device__ __forceinline__ void load_a_f16(v16h& a, const _Float16* __restrict__ F,
                                           int ld, int rowBase, int kBase, int lane) {
  const int m  = lane & 15;
  const int hi = lane >> 4;
  const _Float16* row = F + (size_t)(rowBase + m) * ld + kBase;
  const int k0 = hi ? 16 : 0;
  const int k1 = hi ? 24 : 8;
#pragma unroll
  for (int i = 0; i < 8; ++i) a[i] = row[k0 + i];
#pragma unroll
  for (int i = 0; i < 8; ++i) a[8 + i] = row[k1 + i];
}

// ---- B-operand (32x16, K x N) loaders -------------------------------------
// Layout: lanes 0-15 hold K 0-15 of column n=lane; lanes 16-31 hold K 16-31.
// Source G stores B's columns as contiguous rows: B(k,n) = G[rowBase+n][kBase+k].
__device__ __forceinline__ void load_b_f16rows(v16h& b, const _Float16* __restrict__ G,
                                               int ld, int rowBase, int kBase, int lane) {
  const _Float16* p = G + (size_t)(rowBase + (lane & 15)) * ld + kBase + (lane >> 4) * 16;
#pragma unroll
  for (int i = 0; i < 16; ++i) b[i] = p[i];
}

// B from an f32 row-major [K][Nout] weight matrix (strided; W is tiny/L2-hot).
__device__ __forceinline__ void load_b_f32cols(v16h& b, const float* __restrict__ W,
                                               int ld, int kBase, int nBase, int lane) {
  const float* p = W + (size_t)(kBase + (lane >> 4) * 16) * ld + nBase + (lane & 15);
#pragma unroll
  for (int i = 0; i < 16; ++i) b[i] = (_Float16)p[(size_t)i * ld];
}

// ---------------------------------------------------------------------------
// out_f16 = f16(X[M,K]) @ f16(W[K,Nout]) + bias.
// transposeOut=0: out row-major [M][Nout].  transposeOut=1: out [Nout][M]
// (used for h^T so that the PV B-operand is contiguous along tokens).
// ---------------------------------------------------------------------------
__global__ void __launch_bounds__(128)
proj_gemm_f16(const float* __restrict__ X, const float* __restrict__ W,
              const float* __restrict__ bias, _Float16* __restrict__ out,
              int M, int K, int Nout, int transposeOut) {
  const int lane = threadIdx.x & 31;
  const int wave = threadIdx.x >> 5;
  const int rowBase = blockIdx.x * 16;
  const int nBase = (blockIdx.y * 4 + wave) * 16;
  if (nBase >= Nout) return;  // uniform per wave

  v8f acc = zero_v8f();
  for (int k = 0; k < K; k += 32) {
    v16h a, b;
    load_a_f32(a, X, K, rowBase, k, lane);
    load_b_f32cols(b, W, Nout, k, nBase, lane);
    acc = wmma_f16(a, b, acc);
  }

  const int hi = lane >> 4;
  const int n = nBase + (lane & 15);
  const float bs = bias[n];
  if (!transposeOut) {
#pragma unroll
    for (int r = 0; r < 8; ++r)
      out[(size_t)(rowBase + hi * 8 + r) * Nout + n] = (_Float16)(acc[r] + bs);
  } else {
    // C/D layout: this lane holds column n, 8 consecutive rows -> packed b128.
    v8h o;
#pragma unroll
    for (int r = 0; r < 8; ++r) o[r] = (_Float16)(acc[r] + bs);
    *(v8h*)(out + (size_t)n * M + rowBase + hi * 8) = o;
  }
}

// ---------------------------------------------------------------------------
// Flash attention. Grid = (N/16 query tiles, B). Block = 128 (4 waves).
// The shared 32-key g tile is double-buffered in LDS via async loads; each
// wave owns 128 value channels and redundantly computes the 16x32 score tile
// + online softmax (no inter-wave sync on the softmax state).
// ---------------------------------------------------------------------------
__global__ void __launch_bounds__(128)
flash_attn_f16(const _Float16* __restrict__ f, const _Float16* __restrict__ g,
               const _Float16* __restrict__ hT, const float* __restrict__ X,
               const float* __restrict__ gamma, float* __restrict__ out,
               int Ntok, int C) {
  __shared__ __align__(16) _Float16 Gst[2][32][64];   // double-buffered key tile (8KB)
  __shared__ __align__(16) _Float16 Plds[4][16][32];  // per-wave P staging (4KB)

  const int tid = threadIdx.x;
  const int lane = tid & 31;
  const int wave = tid >> 5;
  const int hi = lane >> 4;
  const int b = blockIdx.y;
  const int qBase = blockIdx.x * 16;
  const int chBase = wave * 128;

  const _Float16* fb = f + (size_t)b * Ntok * 64;
  const _Float16* gb = g + (size_t)b * Ntok * 64;
  const _Float16* hb = hT + (size_t)b * (size_t)C * Ntok;
  const float* Xb = X + (size_t)b * Ntok * C;
  float* outb = out + (size_t)b * Ntok * C;

  // Query tile A-operands, K split 0..31 / 32..63 (live across the whole loop).
  v16h af0, af1;
  load_a_f16(af0, fb, 64, qBase, 0, lane);
  load_a_f16(af1, fb, 64, qBase, 32, lane);

  float mrow[8], lrow[8];
  v8f acc[8];
#pragma unroll
  for (int r = 0; r < 8; ++r) { mrow[r] = -INFINITY; lrow[r] = 0.f; }
#pragma unroll
  for (int t = 0; t < 8; ++t) acc[t] = zero_v8f();

  // Prologue: stage key tile 0.
  async_copy_4k(gb, &Gst[0][0][0], tid);
  int par = 0;

  for (int kt = 0; kt < Ntok; kt += 32) {
    // Kick off the next tile's async copy, then wait for the current one
    // (per-wave ASYNCcnt; loads complete in order -> leave 2 newest in flight).
    if (kt + 32 < Ntok) {
      async_copy_4k(gb + (size_t)(kt + 32) * 64, &Gst[par ^ 1][0][0], tid);
      asm volatile("s_wait_asynccnt 2" ::: "memory");
    } else {
      asm volatile("s_wait_asynccnt 0" ::: "memory");
    }
    __syncthreads();  // all waves' portions of Gst[par] are now resident

    // --- scores: 16 queries x 32 keys, K=64 -> 4 WMMAs (B from LDS) ---------
    const _Float16* Gt = &Gst[par][0][0];
    v16h bg;
    v8f s0 = zero_v8f(), s1 = zero_v8f();
    load_b_f16rows(bg, Gt, 64, 0, 0, lane);    s0 = wmma_f16(af0, bg, s0);
    load_b_f16rows(bg, Gt, 64, 0, 32, lane);   s0 = wmma_f16(af1, bg, s0);
    load_b_f16rows(bg, Gt, 64, 16, 0, lane);   s1 = wmma_f16(af0, bg, s1);
    load_b_f16rows(bg, Gt, 64, 16, 32, lane);  s1 = wmma_f16(af1, bg, s1);

    __syncthreads();  // done reading Gst[par]; next iter may overwrite it

    // --- online softmax. Row m = hi*8 + r lives at vgpr r across 16 lanes,
    //     identical mapping to the O accumulators -> rescale is lane-local.
    float scale[8];
#pragma unroll
    for (int r = 0; r < 8; ++r) {
      float tm = fmaxf(s0[r], s1[r]);
#pragma unroll
      for (int off = 1; off < 16; off <<= 1) tm = fmaxf(tm, __shfl_xor(tm, off, 32));
      const float mnew = fmaxf(mrow[r], tm);
      scale[r] = exp2f((mrow[r] - mnew) * LOG2E);
      const float p0 = exp2f((s0[r] - mnew) * LOG2E);
      const float p1 = exp2f((s1[r] - mnew) * LOG2E);
      float rs = p0 + p1;
#pragma unroll
      for (int off = 1; off < 16; off <<= 1) rs += __shfl_xor(rs, off, 32);
      lrow[r] = lrow[r] * scale[r] + rs;
      mrow[r] = mnew;
      Plds[wave][hi * 8 + r][lane & 15] = (_Float16)p0;
      Plds[wave][hi * 8 + r][16 + (lane & 15)] = (_Float16)p1;
    }
#pragma unroll
    for (int t = 0; t < 8; ++t)
#pragma unroll
      for (int r = 0; r < 8; ++r) acc[t][r] *= scale[r];

    // --- P back in A layout (per-wave private LDS; same-wave DS ordered) ----
    v16h pa;
    {
      const _Float16* prow = &Plds[wave][lane & 15][0];
      const int k0 = hi ? 16 : 0, k1 = hi ? 24 : 8;
#pragma unroll
      for (int i = 0; i < 8; ++i) pa[i] = prow[k0 + i];
#pragma unroll
      for (int i = 0; i < 8; ++i) pa[8 + i] = prow[k1 + i];
    }

    if (kt + 32 < Ntok)
      __builtin_prefetch(hb + (size_t)chBase * Ntok + kt + 32, 0, 1);

    // --- PV: 8 channel sub-tiles of 16 -> 8 WMMAs ---------------------------
#pragma unroll
    for (int t = 0; t < 8; ++t) {
      v16h bh;
      load_b_f16rows(bh, hb, Ntok, chBase + t * 16, kt, lane);
      acc[t] = wmma_f16(pa, bh, acc[t]);
    }
    par ^= 1;
  }

  // --- epilogue: normalize, gamma * o + x ----------------------------------
  const float gm = gamma[0];
  float inv[8];
#pragma unroll
  for (int r = 0; r < 8; ++r) inv[r] = (lrow[r] > 0.f) ? 1.f / lrow[r] : 0.f;
#pragma unroll
  for (int t = 0; t < 8; ++t) {
    const int n = chBase + t * 16 + (lane & 15);
#pragma unroll
    for (int r = 0; r < 8; ++r) {
      const int tok = qBase + hi * 8 + r;
      outb[(size_t)tok * C + n] = gm * acc[t][r] * inv[r] + Xb[(size_t)tok * C + n];
    }
  }
}

// ---------------------------------------------------------------------------
extern "C" void kernel_launch(void* const* d_in, const int* in_sizes, int n_in,
                              void* d_out, int out_size, void* d_ws, size_t ws_size,
                              hipStream_t stream) {
  (void)in_sizes; (void)n_in; (void)out_size; (void)ws_size;

  const int B = 4, Ntok = 64 * 64, C = 512, Dqk = 64;
  const int M = B * Ntok;  // 16384 tokens total

  const float* X     = (const float*)d_in[0];
  const float* Wq    = (const float*)d_in[1];
  const float* bq    = (const float*)d_in[2];
  const float* Wk    = (const float*)d_in[3];
  const float* bk    = (const float*)d_in[4];
  const float* Wv    = (const float*)d_in[5];
  const float* bv    = (const float*)d_in[6];
  const float* gamma = (const float*)d_in[7];
  float* out = (float*)d_out;

  // Workspace: f [M][64] f16 | g [M][64] f16 | hT per batch [C][Ntok] f16 = 20 MB
  _Float16* fbuf  = (_Float16*)d_ws;
  _Float16* gbuf  = fbuf + (size_t)M * Dqk;
  _Float16* hTbuf = gbuf + (size_t)M * Dqk;

  // Q/K projections over all batches at once (X is [B*N, C] row-major).
  proj_gemm_f16<<<dim3(M / 16, 1), 128, 0, stream>>>(X, Wq, bq, fbuf, M, C, Dqk, 0);
  proj_gemm_f16<<<dim3(M / 16, 1), 128, 0, stream>>>(X, Wk, bk, gbuf, M, C, Dqk, 0);

  // V projection per batch, stored transposed [C][Ntok].
  for (int b = 0; b < B; ++b) {
    proj_gemm_f16<<<dim3(Ntok / 16, C / 64), 128, 0, stream>>>(
        X + (size_t)b * Ntok * C, Wv, bv, hTbuf + (size_t)b * C * Ntok,
        Ntok, C, C, 1);
  }

  flash_attn_f16<<<dim3(Ntok / 16, B), 128, 0, stream>>>(
      fbuf, gbuf, hTbuf, X, gamma, out, Ntok, C);
}
